// LSTMCell_27144193310893
// MI455X (gfx1250) — compile-verified, matrix operand order checked by reading, not run on hardware
//
#include <hip/hip_runtime.h>

typedef __attribute__((ext_vector_type(16))) __bf16 v16bf;
typedef __attribute__((ext_vector_type(8)))  float  v8f;
typedef __attribute__((ext_vector_type(4)))  int    int4v;

#define AS_G __attribute__((address_space(1)))
#define AS_L __attribute__((address_space(3)))

#define B_SZ 16384
#define H_SZ 1024
#define K_TOT 2048
#define BM 128
#define BN 32
#define BK 32
#define NKT (K_TOT / BK)   // 64

// bf16 workspace layout in d_ws:
//   Wbf: [32 colTiles][64 kt][4096 ushorts]  = 16 MB   (fragment-ready)
//   Xbf: [128 rowTiles][64 kt][4096 ushorts] = 64 MB   (fragment-ready)
#define WBF_ELEMS ((size_t)32 * 64 * 4096)    // 8,388,608 ushorts
#define XBF_ELEMS ((size_t)128 * 64 * 4096)   // 33,554,432 ushorts

#if __has_builtin(__builtin_amdgcn_global_load_async_to_lds_b128) && \
    __has_builtin(__builtin_amdgcn_s_wait_asynccnt)
#define HAVE_ASYNC_LDS 1
#else
#define HAVE_ASYNC_LDS 0
#endif

__device__ __forceinline__ unsigned short f2bf(float f) {
    unsigned int u = __float_as_uint(f);
    u += 0x7FFFu + ((u >> 16) & 1u);          // round-to-nearest-even
    return (unsigned short)(u >> 16);
}
__device__ __forceinline__ float fast_sigmoid(float x) {
    return 1.0f / (1.0f + __expf(-x));
}
__device__ __forceinline__ float fast_tanh(float x) {
    float t = 1.0f - 2.0f / (__expf(2.0f * fabsf(x)) + 1.0f);
    return copysignf(t, x);
}

// fragment-ready element mapping within one 128(rows)x32(K) tile:
// lane l(0..15) row/col=l holds K{0..7,16..23}; lane l+16 holds K{8..15,24..31}
__device__ __forceinline__ int frag_off(int r16sub, int rl, int k0) {
    int hi  = (k0 >> 3) & 1;
    int idx = (k0 & 7) + ((k0 >> 4) << 3);
    return ((r16sub * 32) + rl + (hi << 4)) * 16 + idx;
}

// ---------------- pre-pass: f32 -> bf16 fragment-ready pack ----------------
__global__ __launch_bounds__(256) void pack_x_kernel(
    const float* __restrict__ input, const float* __restrict__ h_prev,
    unsigned short* __restrict__ Xbf)
{
    int c    = blockIdx.x * 256 + threadIdx.x;   // 8,388,608 float4-chunks
    int kIdx = c & 511;                          // 512 chunks per row
    int rg   = c >> 9;
    int kg0  = kIdx * 4;
    const float* src = (kg0 < H_SZ) ? (input  + (size_t)rg * H_SZ + kg0)
                                    : (h_prev + (size_t)rg * H_SZ + (kg0 - H_SZ));
    float4 v = *(const float4*)src;
    int rowTile = rg >> 7, r = rg & 127;
    int kt = kg0 >> 5, k0 = kg0 & 31;
    size_t off = (size_t)(rowTile * 64 + kt) * 4096 + frag_off(r >> 4, r & 15, k0);
    ushort4 b; b.x = f2bf(v.x); b.y = f2bf(v.y); b.z = f2bf(v.z); b.w = f2bf(v.w);
    *(ushort4*)(Xbf + off) = b;
}

__global__ __launch_bounds__(256) void pack_w_kernel(
    const float* __restrict__ W_ih, const float* __restrict__ W_hh,
    unsigned short* __restrict__ Wbf)
{
    int c    = blockIdx.x * 256 + threadIdx.x;   // 2,097,152 float4-chunks
    int kIdx = c & 511;
    int n    = c >> 9;                           // 0..4095 output neuron
    int kg0  = kIdx * 4;
    const float* src = (kg0 < H_SZ) ? (W_ih + (size_t)n * H_SZ + kg0)
                                    : (W_hh + (size_t)n * H_SZ + (kg0 - H_SZ));
    float4 v = *(const float4*)src;
    int gg = n >> 10;            // gate
    int colGlobal = n & 1023;
    int colTile = colGlobal >> 5, cc = colGlobal & 31;
    int kt = kg0 >> 5, k0 = kg0 & 31;
    size_t off = (size_t)(colTile * 64 + kt) * 4096 +
                 frag_off(gg * 2 + (cc >> 4), cc & 15, k0);
    ushort4 b; b.x = f2bf(v.x); b.y = f2bf(v.y); b.z = f2bf(v.z); b.w = f2bf(v.w);
    *(ushort4*)(Wbf + off) = b;
}

// ---------------- main fused GEMM + gates ----------------
__global__ __launch_bounds__(256) void lstm_cell_kernel(
    const unsigned short* __restrict__ Xbf, const unsigned short* __restrict__ Wbf,
    const float* __restrict__ c_prev, const float* __restrict__ b_ih,
    const float* __restrict__ b_hh, float* __restrict__ h_out,
    float* __restrict__ c_out)
{
    // 64 KB LDS: double-buffered bf16 staging (32 KB) aliased with f32 gate
    // tile gLds (64 KB) used after the K loop.
    __shared__ __align__(32) unsigned char smem[65536];
    unsigned short* Xs = (unsigned short*)smem;            // [2][4096] bf16
    unsigned short* Ws = (unsigned short*)(smem + 16384);  // [2][4096] bf16
    float* gLds = (float*)smem;                            // [4][128][32] f32

    const int tid   = threadIdx.x;
    const int lane  = tid & 31;
    const int wave  = tid >> 5;
    const int gate  = wave & 3;
    const int mhalf = wave >> 2;

    const int rowTile = blockIdx.y;            // 0..127
    const int colTile = blockIdx.x;            // 0..31
    const int rowBase = rowTile * BM;
    const int colBase = colTile * BN;

    // per-thread staging slice: 32 bytes (16 ushorts) of an 8 KB tile
    const int slice = tid * 16;

    const unsigned short* xTileBase = Xbf + (size_t)rowTile * 64 * 4096 + slice;
    const unsigned short* wTileBase = Wbf + (size_t)colTile * 64 * 4096 + slice;

#if HAVE_ASYNC_LDS
    auto stage_async = [&](int buf, int kt) {
        AS_G int4v* xs = (AS_G int4v*)(xTileBase + (size_t)kt * 4096);
        AS_G int4v* ws = (AS_G int4v*)(wTileBase + (size_t)kt * 4096);
        AS_L int4v* xd = (AS_L int4v*)&Xs[buf * 4096 + slice];
        AS_L int4v* wd = (AS_L int4v*)&Ws[buf * 4096 + slice];
        __builtin_amdgcn_global_load_async_to_lds_b128(xs,     xd,     0, 0);
        __builtin_amdgcn_global_load_async_to_lds_b128(xs + 1, xd + 1, 0, 0);
        __builtin_amdgcn_global_load_async_to_lds_b128(ws,     wd,     0, 0);
        __builtin_amdgcn_global_load_async_to_lds_b128(ws + 1, wd + 1, 0, 0);
    };
#endif

    v8f zero = {0.f, 0.f, 0.f, 0.f, 0.f, 0.f, 0.f, 0.f};
    v8f acc[4][2];
#pragma unroll
    for (int t = 0; t < 4; ++t)
#pragma unroll
        for (int u = 0; u < 2; ++u) acc[t][u] = zero;

    // prologue: stage K-step 0 into buffer 0
#if HAVE_ASYNC_LDS
    stage_async(0, 0);
#else
    {
        uint4 x0 = *(const uint4*)xTileBase;
        uint4 x1 = *(const uint4*)(xTileBase + 8);
        uint4 w0 = *(const uint4*)wTileBase;
        uint4 w1 = *(const uint4*)(wTileBase + 8);
        *(uint4*)&Xs[slice]     = x0; *(uint4*)&Xs[slice + 8] = x1;
        *(uint4*)&Ws[slice]     = w0; *(uint4*)&Ws[slice + 8] = w1;
    }
#endif

    for (int kt = 0; kt < NKT; ++kt) {
#if HAVE_ASYNC_LDS
        __builtin_amdgcn_s_wait_asynccnt(0);   // staged data visible before barrier
#endif
        __syncthreads();
        const int cur = kt & 1;
        const bool hasNext = (kt + 1 < NKT);

#if HAVE_ASYNC_LDS
        if (hasNext) stage_async(cur ^ 1, kt + 1);
#else
        uint4 x0, x1, w0, w1;
        if (hasNext) {
            const unsigned short* xs = xTileBase + (size_t)(kt + 1) * 4096;
            const unsigned short* ws = wTileBase + (size_t)(kt + 1) * 4096;
            x0 = *(const uint4*)xs; x1 = *(const uint4*)(xs + 8);
            w0 = *(const uint4*)ws; w1 = *(const uint4*)(ws + 8);
        }
#endif

        v16bf afrag[4];
#pragma unroll
        for (int t = 0; t < 4; ++t)
            afrag[t] = *reinterpret_cast<const v16bf*>(
                &Xs[cur * 4096 + ((mhalf * 4 + t) * 32 + lane) * 16]);
        v16bf bfrag[2];
#pragma unroll
        for (int u = 0; u < 2; ++u)
            bfrag[u] = *reinterpret_cast<const v16bf*>(
                &Ws[cur * 4096 + ((gate * 2 + u) * 32 + lane) * 16]);

#pragma unroll
        for (int t = 0; t < 4; ++t)
#pragma unroll
            for (int u = 0; u < 2; ++u)
                acc[t][u] = __builtin_amdgcn_wmma_f32_16x16x32_bf16(
                    false, afrag[t], false, bfrag[u], (short)0, acc[t][u],
                    false, false);

#if !HAVE_ASYNC_LDS
        if (hasNext) {
            const int nb = (cur ^ 1) * 4096;
            *(uint4*)&Xs[nb + slice]     = x0; *(uint4*)&Xs[nb + slice + 8] = x1;
            *(uint4*)&Ws[nb + slice]     = w0; *(uint4*)&Ws[nb + slice + 8] = w1;
        }
#endif
    }

    // -------- dump g (+bias) into LDS (aliases staging buffers) ----
    __syncthreads();
    float biasv[2];
#pragma unroll
    for (int u = 0; u < 2; ++u) {
        int n = gate * H_SZ + colBase + u * 16 + (lane & 15);
        biasv[u] = b_ih[n] + b_hh[n];
    }
#pragma unroll
    for (int t = 0; t < 4; ++t) {
        int rbase = mhalf * 64 + t * 16 + ((lane >> 4) << 3);
        int col0  = lane & 15;
#pragma unroll
        for (int u = 0; u < 2; ++u)
#pragma unroll
            for (int v = 0; v < 8; ++v)
                gLds[(gate * 128 + (rbase + v)) * 32 + (u * 16 + col0)] =
                    acc[t][u][v] + biasv[u];
    }
    __syncthreads();

    // -------- fused gate math + state update ----
#pragma unroll
    for (int i = 0; i < 16; ++i) {
        int e   = tid + 256 * i;
        int row = e >> 5;
        int col = e & 31;
        size_t gidx = (size_t)(rowBase + row) * H_SZ + (colBase + col);

        float g0 = gLds[(0 * 128 + row) * 32 + col];
        float g1 = gLds[(1 * 128 + row) * 32 + col];
        float g2 = gLds[(2 * 128 + row) * 32 + col];
        float g3 = gLds[(3 * 128 + row) * 32 + col];

        float ig   = fast_sigmoid(g0);
        float fg   = fast_sigmoid(g1);
        float cand = fast_tanh(g2);
        float og   = fast_sigmoid(g3);

        float cp = c_prev[gidx];
        float cn = cp * fg + ig * cand;
        float hn = fast_tanh(cn) * og;

        h_out[gidx] = hn;
        c_out[gidx] = cn;
    }
}

extern "C" void kernel_launch(void* const* d_in, const int* in_sizes, int n_in,
                              void* d_out, int out_size, void* d_ws, size_t ws_size,
                              hipStream_t stream) {
    const float* input  = (const float*)d_in[0];
    const float* h_prev = (const float*)d_in[1];
    const float* c_prev = (const float*)d_in[2];
    const float* W_ih   = (const float*)d_in[3];
    const float* W_hh   = (const float*)d_in[4];
    const float* b_ih   = (const float*)d_in[5];
    const float* b_hh   = (const float*)d_in[6];

    unsigned short* Wbf = (unsigned short*)d_ws;               // 16 MB
    unsigned short* Xbf = Wbf + WBF_ELEMS;                     // +64 MB (needs 80 MB ws)

    float* h_out = (float*)d_out;
    float* c_out = h_out + (size_t)B_SZ * H_SZ;

    // pre-pass packs (memory-bound, ~250 MB total traffic)
    pack_w_kernel<<<dim3(2097152 / 256), dim3(256), 0, stream>>>(W_ih, W_hh, Wbf);
    pack_x_kernel<<<dim3(8388608 / 256), dim3(256), 0, stream>>>(input, h_prev, Xbf);

    dim3 grid(H_SZ / BN, B_SZ / BM);   // 32 x 128
    lstm_cell_kernel<<<grid, dim3(256), 0, stream>>>(Xbf, Wbf, c_prev, b_ih, b_hh,
                                                     h_out, c_out);
}
